// BitConv2d_51900384805158
// MI455X (gfx1250) — matrix operand order repaired and checked by reading, not compiled, not run on hardware
//
#include <hip/hip_runtime.h>
#include <hip/hip_bf16.h>

typedef __attribute__((ext_vector_type(16))) _Float16 v16h;
typedef __attribute__((ext_vector_type(8)))  _Float16 v8h;
typedef __attribute__((ext_vector_type(8)))  float    v8f;

// Problem constants (match reference)
#define BATCH 32
#define CIN   64
#define COUT  64
#define HH    112
#define WW    112
#define NCH   18            // K chunks of 32 (K = 9*64 = 576)
#define TILE_H 4            // output rows per workgroup
#define LDS_ROWS 6          // TILE_H + 2 halo rows
#define LDS_COLS 114        // WW + 2 halo cols
#define SMEM_HALFS (LDS_ROWS * LDS_COLS * CIN)      // 43776
#define SMEM_BYTES (SMEM_HALFS * 2)                 // 87552 (< 320KB WGP LDS)
#define WFRAG_BYTES (4 * NCH * 32 * 16 * 2)         // 73728
#define BIAS_OFF    WFRAG_BYTES                     // f32 bias follows weights in ws

// -----------------------------------------------------------------------------
// Kernel 1: reconstruct f16 weights from bit-planes directly into WMMA
// A-fragment order: wfrag[cb][c][lane][j] (16 f16 per lane = 32B contiguous).
// A-matrix (16x32 f16) layout per CDNA5 ISA:
//   lane = m + 16*half ; VGPR v holds K pair: K(v,half) = (v<4 ? 2v : 2v+8) + 8*half
// Also computes bias[COUT] in f32.
// -----------------------------------------------------------------------------
__global__ void bitconv_build_weights(const float* __restrict__ pweight,
                                      const float* __restrict__ nweight,
                                      const float* __restrict__ scale,
                                      const float* __restrict__ pbias,
                                      const float* __restrict__ nbias,
                                      const float* __restrict__ biasscale,
                                      const int*   __restrict__ nbits_p,
                                      _Float16* __restrict__ wfrag,
                                      float* __restrict__ bias) {
  const int tid = blockIdx.x * blockDim.x + threadIdx.x;
  const int nbits = *nbits_p;
  const float denom = (float)((1 << nbits) - 1);

  if (tid < COUT * CIN * 9) {
    const int o    = tid / (CIN * 9);
    const int i    = (tid / 9) % CIN;
    const int kpos = tid % 9;

    float v = 0.f;
    const long src = (long)tid * nbits;
    for (int bit = 0; bit < nbits; ++bit) {
      const float e = (float)(1 << (nbits - 1 - bit)) / denom;
      v += (pweight[src + bit] - nweight[src + bit]) * e;
    }
    v *= scale[0];

    const int k   = kpos * CIN + i;
    const int c   = k >> 5;
    const int K32 = k & 31;
    const int m   = o & 15;
    const int cb  = o >> 4;
    const int q    = K32 >> 3;
    const int half = q & 1;
    const int vv   = ((q < 2) ? 0 : 4) + ((K32 & 7) >> 1);
    const int pos  = K32 & 1;
    const int lane = m + 16 * half;
    const int j    = 2 * vv + pos;
    wfrag[(((long)cb * NCH + c) * 32 + lane) * 16 + j] = (_Float16)v;
  }

  if (tid < COUT) {
    float bv = 0.f;
    for (int bit = 0; bit < nbits; ++bit) {
      const float e = (float)(1 << (nbits - 1 - bit)) / denom;
      bv += (pbias[tid * nbits + bit] - nbias[tid * nbits + bit]) * e;
    }
    bias[tid] = bv * biasscale[0];
  }
}

// -----------------------------------------------------------------------------
// Kernel 2: implicit-GEMM 3x3 conv via V_WMMA_F32_16X16X32_F16.
// Workgroup: 256 threads = 8 waves; one (batch, 4-row band); all 64 cout.
//   wave w: cb pair {2*(w&1), 2*(w&1)+1}, output row = w>>1 (0..3).
// Each B fragment (1 KB LDS read) feeds TWO wmma (M=32 per wave), halving LDS
// bandwidth; the odd cb's A fragment comes from global (WGP$-resident 73KB),
// splitting operand traffic across the LDS and VMEM paths. An empty asm
// pointer barrier blocks GVN from merging those loads across the unrolled
// pixel-group loop (which would add ~144 live VGPRs), while leaving them as
// normal loads the scheduler can batch and overlap (unlike volatile).
// LDS: x halo staged as f16 [row][col][ci] (ci innermost, 8-ci packed b128
// stores) -> per-lane B fragment = two 16B chunks (ci0+8*half, +16) (ds_read_b128 x2).
// -----------------------------------------------------------------------------
__global__ void __launch_bounds__(256, 1)
bitconv_wmma(const float* __restrict__ x,
             const _Float16* __restrict__ wfrag,
             const float* __restrict__ bias,
             float* __restrict__ out) {
  extern __shared__ _Float16 smem[];

  const int tid = threadIdx.x;
  const int b   = blockIdx.y;
  const int h0  = blockIdx.x * TILE_H;

  // ---- stage input halo band into LDS as f16, ci innermost ----------------
  // interior: unit = (rr, ci-group of 8, 4-col chunk): 6*8*28 = 1344 units
  const float* xb = x + (long)b * CIN * HH * WW;
  for (int u = tid; u < LDS_ROWS * 8 * 28; u += 256) {
    const int chunk = u % 28;
    const int t     = u / 28;
    const int cig   = t & 7;
    const int rr    = t >> 3;                 // 0..5
    const int row   = h0 - 1 + rr;
    const int colb  = 1 + chunk * 4;
    _Float16* dst = smem + ((long)(rr * LDS_COLS + colb) * CIN + cig * 8);
    if (row >= 0 && row < HH) {
      float4 p[8];
#pragma unroll
      for (int pl = 0; pl < 8; ++pl)
        p[pl] = *(const float4*)(xb + ((long)(cig * 8 + pl) * HH + row) * WW + chunk * 4);
#pragma unroll
      for (int q = 0; q < 4; ++q) {
        v8h hv;
#pragma unroll
        for (int pl = 0; pl < 8; ++pl) hv[pl] = (_Float16)((&p[pl].x)[q]);
        *(v8h*)(dst + (long)q * CIN) = hv;    // ds_store_b128
      }
    } else {
      const v8h z = {};
#pragma unroll
      for (int q = 0; q < 4; ++q) *(v8h*)(dst + (long)q * CIN) = z;
    }
  }
  // halo columns (w = -1 and w = 112) are always zero: 6*8*2 = 96 units
  for (int u = tid; u < LDS_ROWS * 8 * 2; u += 256) {
    const int side = u & 1;
    const int t    = u >> 1;
    const int cig  = t & 7;
    const int rr   = t >> 3;
    const int col  = side ? (LDS_COLS - 1) : 0;
    const v8h z = {};
    *(v8h*)(smem + ((long)(rr * LDS_COLS + col) * CIN + cig * 8)) = z;
  }
  __syncthreads();

  // ---- per-wave setup ------------------------------------------------------
  const int wave = tid >> 5;
  const int lane = tid & 31;
  const int cb0  = (wave & 1) * 2;       // even cb of this wave's pair
  const int cb1  = cb0 + 1;              // odd cb (A streamed from global/WGP$)
  const int r_out = wave >> 1;           // 0..3
  const int half = lane >> 4;
  const int n    = lane & 15;

  // preload the 18 A fragments of the even channel block into registers
  v16h A0[NCH];
#pragma unroll
  for (int c = 0; c < NCH; ++c)
    A0[c] = *(const v16h*)(wfrag + (((long)cb0 * NCH + c) * 32 + lane) * 16);

  // bias per accumulator VGPR: acc[r] holds M = r + 8*half
  float bs0[8], bs1[8];
#pragma unroll
  for (int r = 0; r < 8; ++r) {
    bs0[r] = bias[cb0 * 16 + r + 8 * half];
    bs1[r] = bias[cb1 * 16 + r + 8 * half];
  }

  const int h = h0 + r_out;
  for (int g = 0; g < WW / 16; ++g) {
    const int w0 = g * 16;
    v8f acc0 = {};
    v8f acc1 = {};
#pragma unroll
    for (int c = 0; c < NCH; ++c) {
      const int kpos = c >> 1;
      const int kh = kpos / 3;
      const int kw = kpos % 3;
      const int ci0 = (c & 1) << 5;
      const int rr = r_out + kh;           // input row in LDS band
      const int col = w0 + n + kw;         // padded column, 0..113
      const _Float16* bp = smem + ((long)(rr * LDS_COLS + col) * CIN + ci0 + 8 * half);
      const v8h blo = *(const v8h*)bp;
      const v8h bhi = *(const v8h*)(bp + 16);
      const v16h bf = __builtin_shufflevector(blo, bhi, 0,1,2,3,4,5,6,7,8,9,10,11,12,13,14,15);
      // odd-cb A fragment: normal 32B load through an opaque pointer.
      // asm barrier defeats CSE/LICM across g-iterations (no 144-VGPR hoist),
      // but the load remains schedulable/batchable (proper s_wait_loadcnt<=N).
      const _Float16* wp1 = wfrag + (((long)cb1 * NCH + c) * 32 + lane) * 16;
      asm("" : "+v"(wp1));
      const v16h a1 = *(const v16h*)wp1;
      acc0 = __builtin_amdgcn_wmma_f32_16x16x32_f16(
          false, A0[c], false, bf, (short)0, acc0, false, false);
      acc1 = __builtin_amdgcn_wmma_f32_16x16x32_f16(
          false, a1,    false, bf, (short)0, acc1, false, false);
    }
    // store D (2 x 16 cout x 16 pixels) + bias
    const long base0 = (((long)(b * COUT + cb0 * 16) * HH + h) * WW) + w0 + n;
    const long base1 = (((long)(b * COUT + cb1 * 16) * HH + h) * WW) + w0 + n;
#pragma unroll
    for (int r = 0; r < 8; ++r) {
      out[base0 + (long)(r + 8 * half) * (HH * WW)] = acc0[r] + bs0[r];
      out[base1 + (long)(r + 8 * half) * (HH * WW)] = acc1[r] + bs1[r];
    }
  }
}

// -----------------------------------------------------------------------------
extern "C" void kernel_launch(void* const* d_in, const int* in_sizes, int n_in,
                              void* d_out, int out_size, void* d_ws, size_t ws_size,
                              hipStream_t stream) {
  const float* x          = (const float*)d_in[0];
  const float* pweight    = (const float*)d_in[1];
  const float* nweight    = (const float*)d_in[2];
  const float* scale      = (const float*)d_in[3];
  const float* pbias      = (const float*)d_in[4];
  const float* nbias      = (const float*)d_in[5];
  const float* biasscale  = (const float*)d_in[6];
  const int*   nbits      = (const int*)d_in[7];

  _Float16* wfrag = (_Float16*)d_ws;
  float*    bias  = (float*)((char*)d_ws + BIAS_OFF);
  float*    out   = (float*)d_out;

  // 1) reconstruct weights into WMMA fragment order + bias
  const int nw = COUT * CIN * 9;
  bitconv_build_weights<<<(nw + 255) / 256, 256, 0, stream>>>(
      pweight, nweight, scale, pbias, nbias, biasscale, nbits, wfrag, bias);

  // 2) implicit-GEMM conv: grid = (H/4 bands, batches)
  dim3 grid(HH / TILE_H, BATCH, 1);
  bitconv_wmma<<<grid, 256, SMEM_BYTES, stream>>>(x, wfrag, bias, out);
}
